// DLLinearZeroDiagonal_5480378270214
// MI455X (gfx1250) — matrix operand order repaired and checked by reading, not compiled, not run on hardware
//
#include <hip/hip_runtime.h>

typedef __attribute__((ext_vector_type(16))) _Float16 v16h;
typedef __attribute__((ext_vector_type(8)))  _Float16 v8h;
typedef __attribute__((ext_vector_type(4)))  _Float16 v4h;
typedef __attribute__((ext_vector_type(8)))  float    v8f;
typedef __attribute__((ext_vector_type(4)))  float    v4f;

#define N_DIM 4096
#define LDT   40                    // LDS row stride in halves (32 + 8 pad -> conflict-free)
#define MAT_HALVES (128 * LDT)      // one 128x32 f16 tile in LDS (5120 halves = 10240 B)
#define BUF_HALVES (2 * MAT_HALVES) // A-tile + B-tile per pipeline buffer

// ---------------------------------------------------------------------------
// 1) x (fp32) -> Ah (f16), elementwise, 4 elems/thread
// ---------------------------------------------------------------------------
__global__ __launch_bounds__(256) void cvt_x_f16(const float* __restrict__ x,
                                                 _Float16* __restrict__ ah) {
    size_t i = ((size_t)blockIdx.x * 256 + threadIdx.x) * 4;
    v4f v = *(const v4f*)(x + i);
    v4h h;
    h.x = (_Float16)v.x;
    h.y = (_Float16)v.y;
    h.z = (_Float16)v.z;
    h.w = (_Float16)v.w;
    *(v4h*)(ah + i) = h;
}

// ---------------------------------------------------------------------------
// 2) scatter triangular weights into dense row-major W[n][k] as f16
// ---------------------------------------------------------------------------
__global__ __launch_bounds__(256) void pack_w_f16(const float* __restrict__ uw,
                                                  const float* __restrict__ lw,
                                                  _Float16* __restrict__ wh) {
    const int n = blockIdx.y;
    const int k = blockIdx.x * 256 + threadIdx.x;
    const int tri = (n * (n - 1)) >> 1;     // fits in int for n < 4096
    float v;
    if (k > n)      v = uw[n * (N_DIM - 1) - tri + (k - n - 1)];
    else if (k < n) v = lw[tri + k];
    else            v = 0.0f;
    wh[(size_t)n * N_DIM + k] = (_Float16)v;
}

// ---------------------------------------------------------------------------
// async global -> LDS copy, 16B per lane, tracked by ASYNCcnt.
// ldsAddr is derived from a real pointer into the __shared__ array (ptrtoint
// captures the array so alias analysis keeps the later ds_loads alive; low 32
// bits of the flat address == wave-relative LDS byte address, ISA 10.2).
// ---------------------------------------------------------------------------
__device__ __forceinline__ void async_copy_b128(const _Float16* ldsPtr,
                                                unsigned gByte,
                                                const void* base) {
    unsigned ldsAddr = (unsigned)(size_t)ldsPtr;
    asm volatile("global_load_async_to_lds_b128 %0, %1, %2"
                 :: "v"(ldsAddr), "v"(gByte), "s"(base)
                 : "memory");
}

// ---------------------------------------------------------------------------
// 3) GEMM: out[m][n] = sum_k Ah[m][k] * Wh[n][k] + bias[n]
//    128x128 block tile, 8 waves, each wave: 64(M) x 32(N) = 4x2 WMMA tiles.
//    Double-buffered LDS fed by GLOBAL_LOAD_ASYNC_TO_LDS_B128.
// ---------------------------------------------------------------------------
__global__ __launch_bounds__(256) void gemm_tri_f16(const _Float16* __restrict__ Ah,
                                                    const _Float16* __restrict__ Bh,
                                                    const float* __restrict__ bias,
                                                    float* __restrict__ out) {
    __shared__ _Float16 smem[2 * BUF_HALVES];   // 40960 bytes, 2 pipeline buffers

    const int tid    = threadIdx.x;
    const int lane   = tid & 31;
    const int wid    = tid >> 5;
    const int lane16 = lane & 15;
    const int laneHi = lane >> 4;          // 0 or 1 (half-wave)
    const int waveM  = wid & 1;            // 2 waves along M -> 64 rows each
    const int waveN  = wid >> 1;           // 4 waves along N -> 32 cols each

    const int m0 = blockIdx.y * 128;
    const int n0 = blockIdx.x * 128;

    v8f acc[4][2] = {};                    // fp32 accumulators

    // Per tile: 512 b128 chunks per matrix; this thread owns chunks 2t,2t+1
    // (row = c>>2, ko = (c&3)*8 halves). 4 async ops per thread per tile.
    const int c0 = tid * 2;

    auto issue_tile = [&](int kk, int buf) {
        #pragma unroll
        for (int j = 0; j < 2; ++j) {
            const int c   = c0 + j;
            const int row = c >> 2;
            const int ko  = (c & 3) * 8;
            const _Float16* lA = smem + buf * BUF_HALVES + row * LDT + ko;
            const unsigned gA = (unsigned)(((m0 + row) * N_DIM + kk + ko) * 2);
            async_copy_b128(lA, gA, Ah);
            const _Float16* lB = lA + MAT_HALVES;
            const unsigned gB = (unsigned)(((n0 + row) * N_DIM + kk + ko) * 2);
            async_copy_b128(lB, gB, Bh);
        }
    };

    const int nIter = N_DIM / 32;          // 128 K-steps
    issue_tile(0, 0);                      // prologue: tile 0 -> buf 0

    for (int t = 0; t < nIter; ++t) {
        const int buf = t & 1;
        if (t + 1 < nIter) {
            issue_tile((t + 1) * 32, buf ^ 1);                 // prefetch next tile
            asm volatile("s_wait_asynccnt 0x4" ::: "memory");  // tile t landed
        } else {
            asm volatile("s_wait_asynccnt 0x0" ::: "memory");
        }
        __syncthreads();                   // all waves' async data visible

        const _Float16* As = smem + buf * BUF_HALVES;
        const _Float16* Bs = As + MAT_HALVES;

        // --- A fragments: 16-bit A 16x32 layout (ISA 7.12.2) ---
        // lanes 0-15 (M=lane): K0..7 then K16..23 ; lanes 16-31: K8..15, K24..31
        v16h afrag[4];
        #pragma unroll
        for (int mi = 0; mi < 4; ++mi) {
            const _Float16* p = As + (waveM * 64 + mi * 16 + lane16) * LDT + laneHi * 8;
            union { v16h v; v8h h[2]; } u;
            u.h[0] = *(const v8h*)(p);         // K chunk (8 halves)
            u.h[1] = *(const v8h*)(p + 16);    // K chunk +16
            afrag[mi] = u.v;
        }
        // --- B fragments: column n per lane, same K order as A ---
        // lanes 0-15: K0..7 then K16..23 ; lanes 16-31: K8..15, K24..31
        v16h bfrag[2];
        #pragma unroll
        for (int ni = 0; ni < 2; ++ni) {
            const _Float16* p = Bs + (waveN * 32 + ni * 16 + lane16) * LDT + laneHi * 8;
            union { v16h v; v8h h[2]; } u;
            u.h[0] = *(const v8h*)(p);
            u.h[1] = *(const v8h*)(p + 16);
            bfrag[ni] = u.v;
        }

        #pragma unroll
        for (int mi = 0; mi < 4; ++mi) {
            #pragma unroll
            for (int ni = 0; ni < 2; ++ni) {
                acc[mi][ni] = __builtin_amdgcn_wmma_f32_16x16x32_f16(
                    /*neg_a=*/false, afrag[mi],
                    /*neg_b=*/false, bfrag[ni],
                    /*c_mod=*/(short)0, acc[mi][ni],
                    /*reuse_a=*/false, /*reuse_b=*/false);
            }
        }

        __syncthreads();                   // buf t&1 free before t+1's async issue
    }

    // Epilogue: C/D layout -> VGPR v holds M = v (lanes 0-15) / v+8 (lanes 16-31)
    #pragma unroll
    for (int ni = 0; ni < 2; ++ni) {
        const int n  = n0 + waveN * 32 + ni * 16 + lane16;
        const float bv = bias[n];
        #pragma unroll
        for (int mi = 0; mi < 4; ++mi) {
            const int mbase = m0 + waveM * 64 + mi * 16 + laneHi * 8;
            #pragma unroll
            for (int v = 0; v < 8; ++v) {
                out[(size_t)(mbase + v) * N_DIM + n] = acc[mi][ni][v] + bv;
            }
        }
    }
}

// ---------------------------------------------------------------------------
extern "C" void kernel_launch(void* const* d_in, const int* in_sizes, int n_in,
                              void* d_out, int out_size, void* d_ws, size_t ws_size,
                              hipStream_t stream) {
    (void)in_sizes; (void)n_in; (void)out_size; (void)ws_size;

    const float* x    = (const float*)d_in[0];
    const float* uw   = (const float*)d_in[1];
    const float* lw   = (const float*)d_in[2];
    const float* bias = (const float*)d_in[3];
    float* out = (float*)d_out;

    _Float16* Ah = (_Float16*)d_ws;                                      // 32 MB
    _Float16* Wh = (_Float16*)((char*)d_ws + (size_t)N_DIM * N_DIM * 2); // next 32 MB

    // 1) x -> f16
    cvt_x_f16<<<(N_DIM * (size_t)N_DIM) / (4 * 256), 256, 0, stream>>>(x, Ah);
    // 2) triangular scatter -> dense f16 W[n][k]
    pack_w_f16<<<dim3(N_DIM / 256, N_DIM), 256, 0, stream>>>(uw, lw, Wh);
    // 3) GEMM + bias (f16 operands stay resident in 192 MB L2)
    gemm_tri_f16<<<dim3(N_DIM / 128, N_DIM / 128), 256, 0, stream>>>(Ah, Wh, bias, out);
}
